// MyRNN_15281493639833
// MI455X (gfx1250) — matrix operand order repaired and checked by reading, not compiled, not run on hardware
//
#include <hip/hip_runtime.h>
#include <hip/hip_bf16.h>

// ---------------------------------------------------------------------------
// Two stacked SimpleRNNs (Keras-style), BATCH=4096, T=80, EMB=100, UNITS=256.
//   x    = emb[tokens]                                  [B,T,100]
//   h1_t = tanh(x_t@W1 + b1 + h1_{t-1}@U1)   (seq out)
//   h2_t = tanh(h1_t@W2 + b2 + h2_{t-1}@U2)  (last only)
//   out  = sigmoid(h2_T)                                [B,256] f32
//
// MI455X (CDNA5/gfx1250) strategy:
//  K1: embed + input-projection GEMM (bf16 WMMA) -> xp buffer (bf16, 160MiB ws)
//  K2: layer1 recurrence FUSED with layer2 input projection. U1+W2 resident
//      in LDS as bf16 (128KB+128KB) -- needs CDNA5's 320KB LDS. xp2 overwrites
//      xp1 in place. Per-step activations streamed with
//      GLOBAL_LOAD_ASYNC_TO_LDS_B128 (ASYNCcnt) into a double-buffered tile,
//      overlapping HBM latency with the step's WMMA work.
//  K3: layer2 recurrence, same async staging; final step stores sigmoid f32.
// All matmuls: v_wmma_f32_16x16x32_bf16 with f32 accumulation.
// ---------------------------------------------------------------------------

typedef __attribute__((ext_vector_type(16))) __bf16 v16bf;
typedef __attribute__((ext_vector_type(8)))  __bf16 v8bf;
typedef __attribute__((ext_vector_type(8)))  float  v8f;
typedef int v4i __attribute__((__vector_size__(16)));   // matches async builtin param

#define BATCH 4096
#define T     80
#define EMB   100
#define KPAD  128   // EMB padded to multiple of 32 (WMMA K)
#define H     256   // UNITS
#define NT    16    // N tiles (16 wide) over H
#define KT_H  8     // K tiles (32 deep) over H
#define KT_E  4     // K tiles (32 deep) over KPAD

__device__ __forceinline__ v16bf cat16(v8bf lo, v8bf hi) {
  return __builtin_shufflevector(lo, hi, 0, 1, 2, 3, 4, 5, 6, 7,
                                         8, 9, 10, 11, 12, 13, 14, 15);
}

// B operand (32x16, K x N) fetched from pre-swizzled LDS weights.
// Tile (kt,nt) stored as 512 halves: index = n_local*32 + k_local, so a lane
// (col = lane&15, K-half = lane>>4) reads 16 contiguous halves (2x b128).
__device__ __forceinline__ v16bf load_b(const __bf16* w, int kt, int nt, int lane) {
  const __bf16* p = w + ((kt << 4) + nt) * 512 + (lane & 15) * 32 + ((lane >> 4) << 4);
  v8bf lo = *(const v8bf*)p;
  v8bf hi = *(const v8bf*)(p + 8);
  return cat16(lo, hi);
}

// A operand (16x32, M x K) from row-major LDS staging (row stride in halves).
// Layout: lanes 0-15 = rows, K {0..15}; lanes 16-31 K {16..31} (matches load_b).
__device__ __forceinline__ v16bf load_a(const __bf16* s, int kt, int lane, int stride) {
  const __bf16* p = s + (lane & 15) * stride + (kt << 5) + ((lane >> 4) << 4);
  v8bf lo = *(const v8bf*)p;
  v8bf hi = *(const v8bf*)(p + 8);
  return cat16(lo, hi);
}

__device__ __forceinline__ v8f wmma_bf16(v16bf a, v16bf b, v8f c) {
  return __builtin_amdgcn_wmma_f32_16x16x32_bf16(false, a, false, b,
                                                 (short)0, c, false, false);
}

// Swizzled-LDS destination index for weight element (kg, ng) of a KxH matrix.
__device__ __forceinline__ int wswz(int kg, int ng) {
  return ((kg >> 5) * NT + (ng >> 4)) * 512 + (ng & 15) * 32 + (kg & 31);
}

// Async DMA: 16 bytes global -> LDS, tracked by ASYNCcnt (no VGPR round trip).
// Generic->AS casts done via integer round-trip (low 32 bits of a generic LDS
// pointer are the LDS byte offset per the aperture mapping).
__device__ __forceinline__ void async_ld_b128(const void* g, void* l) {
  __builtin_amdgcn_global_load_async_to_lds_b128(
      (__attribute__((address_space(1))) v4i*)(unsigned long long)(size_t)g,
      (__attribute__((address_space(3))) v4i*)(unsigned)(size_t)l,
      0, 0);
}

// Issue the 16x256 bf16 activation tile for timestep tt into sXP buffer `buf`.
// 256 threads x 32 bytes each (2x b128): thread -> (row = tid>>4, 16 halves).
__device__ __forceinline__ void stage_async(const __bf16* xp, __bf16* sXPb,
                                            int rowb, int tt, int tid) {
  const int r = tid >> 4;
  const int c = (tid & 15) << 4;  // half index within row
  const __bf16* g = xp + ((size_t)(rowb + r) * T + tt) * H + c;
  __bf16* l = sXPb + r * H + c;
  async_ld_b128(g, l);
  async_ld_b128(g + 8, l + 8);
}

// ---------------------------------------------------------------------------
// K1: xp[b*T+t, :] = emb[tok[b,t]] @ W1 + b1   (stored bf16)
// grid: (B*T)/128 blocks of 256 threads; each wave owns a 16-row M tile.
// ---------------------------------------------------------------------------
__global__ void rnn_embed_proj1(const int* __restrict__ tok,
                                const float* __restrict__ emb,
                                const float* __restrict__ W1,
                                const float* __restrict__ b1,
                                __bf16* __restrict__ xp) {
  extern __shared__ __align__(16) char smem[];
  __bf16* sW = (__bf16*)smem;                    // KPAD x H swizzled (64KB)
  __bf16* sX = (__bf16*)(smem + KPAD * H * 2);   // 8 waves x 16 x KPAD (32KB)
  const int tid = threadIdx.x, lane = tid & 31, w = tid >> 5;

  // Load W1 (100x256 f32) -> padded swizzled bf16 LDS.
  for (int i = tid; i < KPAD * H; i += 256) {
    int kg = i >> 8, ng = i & 255;
    float v = (kg < EMB) ? W1[kg * H + ng] : 0.f;
    sW[wswz(kg, ng)] = (__bf16)v;
  }

  // Gather this wave's 16 embedding rows, pad K to 128.
  const int rowbase = blockIdx.x * 128 + w * 16;
  __bf16* sx = sX + w * 16 * KPAD;
  for (int i = lane; i < 16 * KPAD; i += 32) {
    int r = i >> 7, k = i & 127;
    int tk = tok[rowbase + r];
    float v = (k < EMB) ? emb[tk * EMB + k] : 0.f;
    sx[r * KPAD + k] = (__bf16)v;
  }
  __syncthreads();

  v16bf a[KT_E];
#pragma unroll
  for (int kt = 0; kt < KT_E; ++kt) a[kt] = load_a(sx, kt, lane, KPAD);

  const int mofs = (lane >> 4) << 3;
  const int nl = lane & 15;
  for (int nt = 0; nt < NT; ++nt) {
    v8f acc = {};
#pragma unroll
    for (int kt = 0; kt < KT_E; ++kt)
      acc = wmma_bf16(a[kt], load_b(sW, kt, nt, lane), acc);
    int ng = nt * 16 + nl;
    float bias = b1[ng];
#pragma unroll
    for (int v = 0; v < 8; ++v) {
      int row = rowbase + v + mofs;   // C/D layout: m = vgpr + 8*(lane>=16)
      xp[(size_t)row * H + ng] = (__bf16)(acc[v] + bias);
    }
  }
}

// ---------------------------------------------------------------------------
// K2: for t: h1 = tanh(xp1_t + h1@U1); xp_t <- h1@W2 + b2  (in place)
// grid: B/16 blocks; LDS = U1(128K) + W2(128K) + h dbl-buf + xp dbl-buf.
// ---------------------------------------------------------------------------
__global__ void rnn_layer1(__bf16* __restrict__ xp,
                           const float* __restrict__ U1,
                           const float* __restrict__ W2,
                           const float* __restrict__ b2) {
  extern __shared__ __align__(16) char smem[];
  __bf16* sU  = (__bf16*)smem;       // 65536 halves
  __bf16* sW  = sU + H * H;          // 65536 halves
  __bf16* sH  = sW + H * H;          // 2 x 16 x 256
  __bf16* sXP = sH + 2 * 16 * H;     // 2 x 16 x 256 (async double buffer)
  const int tid = threadIdx.x, lane = tid & 31, w = tid >> 5;
  const int rowb = blockIdx.x * 16;

  for (int i = tid; i < H * H; i += 256) {
    int kg = i >> 8, ng = i & 255;
    int dst = wswz(kg, ng);
    sU[dst] = (__bf16)U1[i];
    sW[dst] = (__bf16)W2[i];
  }
  for (int i = tid; i < 16 * H; i += 256) sH[i] = (__bf16)0.f;

  stage_async(xp, sXP, rowb, 0, tid);   // prologue: t=0 tile -> buffer 0

  const int nt0 = w * 2;               // each wave owns 2 N tiles
  const int mofs = (lane >> 4) << 3;
  const int nl = lane & 15;
  int cur = 0;

  for (int t = 0; t < T; ++t) {
    const int xb = t & 1;
    __builtin_amdgcn_s_wait_asynccnt(0);
    __syncthreads();                    // sXP[xb] ready everywhere; sH settled
    if (t + 1 < T) stage_async(xp, sXP + (xb ^ 1) * 16 * H, rowb, t + 1, tid);
    const __bf16* sxp = sXP + xb * 16 * H;

    // h_new = tanh(xp + h @ U1)
    v16bf a[KT_H];
#pragma unroll
    for (int kt = 0; kt < KT_H; ++kt) a[kt] = load_a(sH + cur * 16 * H, kt, lane, H);
#pragma unroll
    for (int j = 0; j < 2; ++j) {
      int nt = nt0 + j;
      v8f acc = {};
#pragma unroll
      for (int kt = 0; kt < KT_H; ++kt)
        acc = wmma_bf16(a[kt], load_b(sU, kt, nt, lane), acc);
      int ng = nt * 16 + nl;
#pragma unroll
      for (int v = 0; v < 8; ++v) {
        int m = v + mofs;
        float x = acc[v] + (float)sxp[m * H + ng];
        sH[(cur ^ 1) * 16 * H + m * H + ng] = (__bf16)tanhf(x);
      }
    }
    __syncthreads();

    // xp2_t = h_new @ W2 + b2  -> overwrite xp slot (already consumed above).
#pragma unroll
    for (int kt = 0; kt < KT_H; ++kt) a[kt] = load_a(sH + (cur ^ 1) * 16 * H, kt, lane, H);
#pragma unroll
    for (int j = 0; j < 2; ++j) {
      int nt = nt0 + j;
      v8f acc = {};
#pragma unroll
      for (int kt = 0; kt < KT_H; ++kt)
        acc = wmma_bf16(a[kt], load_b(sW, kt, nt, lane), acc);
      int ng = nt * 16 + nl;
      float bias = b2[ng];
#pragma unroll
      for (int v = 0; v < 8; ++v) {
        int m = v + mofs;
        xp[((size_t)(rowb + m) * T + t) * H + ng] = (__bf16)(acc[v] + bias);
      }
    }
    cur ^= 1;
  }
}

// ---------------------------------------------------------------------------
// K3: for t: h2 = tanh(xp2_t + h2@U2); out = sigmoid(h2_T) (f32)
// ---------------------------------------------------------------------------
__global__ void rnn_layer2(const __bf16* __restrict__ xp,
                           const float* __restrict__ U2,
                           float* __restrict__ out) {
  extern __shared__ __align__(16) char smem[];
  __bf16* sU  = (__bf16*)smem;       // 65536 halves
  __bf16* sH  = sU + H * H;          // 2 x 16 x 256
  __bf16* sXP = sH + 2 * 16 * H;     // 2 x 16 x 256 (async double buffer)
  const int tid = threadIdx.x, lane = tid & 31, w = tid >> 5;
  const int rowb = blockIdx.x * 16;

  for (int i = tid; i < H * H; i += 256) {
    int kg = i >> 8, ng = i & 255;
    sU[wswz(kg, ng)] = (__bf16)U2[i];
  }
  for (int i = tid; i < 16 * H; i += 256) sH[i] = (__bf16)0.f;

  stage_async(xp, sXP, rowb, 0, tid);

  const int nt0 = w * 2;
  const int mofs = (lane >> 4) << 3;
  const int nl = lane & 15;
  int cur = 0;

  for (int t = 0; t < T; ++t) {
    const int xb = t & 1;
    __builtin_amdgcn_s_wait_asynccnt(0);
    __syncthreads();
    if (t + 1 < T) stage_async(xp, sXP + (xb ^ 1) * 16 * H, rowb, t + 1, tid);
    const __bf16* sxp = sXP + xb * 16 * H;

    v16bf a[KT_H];
#pragma unroll
    for (int kt = 0; kt < KT_H; ++kt) a[kt] = load_a(sH + cur * 16 * H, kt, lane, H);
#pragma unroll
    for (int j = 0; j < 2; ++j) {
      int nt = nt0 + j;
      v8f acc = {};
#pragma unroll
      for (int kt = 0; kt < KT_H; ++kt)
        acc = wmma_bf16(a[kt], load_b(sU, kt, nt, lane), acc);
      int ng = nt * 16 + nl;
#pragma unroll
      for (int v = 0; v < 8; ++v) {
        int m = v + mofs;
        float h = tanhf(acc[v] + (float)sxp[m * H + ng]);
        sH[(cur ^ 1) * 16 * H + m * H + ng] = (__bf16)h;
        if (t == T - 1) {
          out[(size_t)(rowb + m) * H + ng] = 1.f / (1.f + __expf(-h));
        }
      }
    }
    __syncthreads();
    cur ^= 1;
  }
}

// ---------------------------------------------------------------------------
extern "C" void kernel_launch(void* const* d_in, const int* in_sizes, int n_in,
                              void* d_out, int out_size, void* d_ws, size_t ws_size,
                              hipStream_t stream) {
  (void)in_sizes; (void)n_in; (void)out_size; (void)ws_size;
  const int*   tok = (const int*)d_in[0];
  const float* emb = (const float*)d_in[1];
  const float* W1  = (const float*)d_in[2];
  const float* U1  = (const float*)d_in[3];
  const float* b1  = (const float*)d_in[4];
  const float* W2  = (const float*)d_in[5];
  const float* U2  = (const float*)d_in[6];
  const float* b2  = (const float*)d_in[7];
  // d_in[8], d_in[9] (Wd, bd) intentionally unused: reference discards s.

  __bf16* xp = (__bf16*)d_ws;   // BATCH*T*H bf16 = 160 MiB scratch, reused in place
  float*  out = (float*)d_out;  // [BATCH, H] f32

  const size_t smA = (size_t)(KPAD * H + 8 * 16 * KPAD) * 2;            //  96 KiB
  const size_t smB = (size_t)(2 * H * H + 2 * 16 * H + 2 * 16 * H) * 2; // 288 KiB
  const size_t smC = (size_t)(H * H + 2 * 16 * H + 2 * 16 * H) * 2;     // 160 KiB

  rnn_embed_proj1<<<(BATCH * T) / 128, 256, smA, stream>>>(tok, emb, W1, b1, xp);
  rnn_layer1<<<BATCH / 16, 256, smB, stream>>>(xp, U1, W2, b2);
  rnn_layer2<<<BATCH / 16, 256, smC, stream>>>(xp, U2, out);
}
